// ConvolutionalMultiheadAttention_65781719105793
// MI455X (gfx1250) — matrix operand order repaired and checked by reading, not compiled, not run on hardware
//
#include <hip/hip_runtime.h>
#include <hip/hip_bf16.h>

typedef __attribute__((ext_vector_type(16))) _Float16 v16h;
typedef __attribute__((ext_vector_type(8)))  _Float16 v8h;
typedef __attribute__((ext_vector_type(8)))  float    v8f;
typedef __attribute__((ext_vector_type(4)))  float    v4f;

#define BM 128
#define BN 64
#define BK 32
#define LDP 40   // padded LDS row stride (f16): 32 + 8 -> conflict-free, 16B aligned rows

// ---------------------------------------------------------------------------
// Shared WMMA GEMM core: C[BMxBN] += A[BMxK] * B^T  (B stored N-major [N][K]
// unless BTRANS, in which case B is K-major [K][N] and is transposed into LDS).
// 256 threads = 8 waves; wave (wm 0..3, wn 0..1) owns a 32x32 region = 2x2 WMMA.
// ---------------------------------------------------------------------------
template<bool AF32, bool BTRANS>
__device__ __forceinline__ void gemm_tile(const void* Agv, int lda,
                                          const _Float16* Bg, int ldb,
                                          int K, int bm, int bn,
                                          _Float16* sA, _Float16* sB,
                                          v8f c[2][2]) {
  const int t    = threadIdx.x;
  const int lane = t & 31;
  const int wave = t >> 5;
  const int wm   = wave >> 1;    // 0..3
  const int wn   = wave & 1;     // 0..1
  const int m    = lane & 15;
  const int hh   = lane >> 4;

  for (int k0 = 0; k0 < K; k0 += BK) {
    __syncthreads();
    // ---- stage A tile (BM x BK) into LDS, converting to f16 if needed ----
    if constexpr (AF32) {
      const float* Ag = (const float*)Agv;
      int row = t >> 3;            // 0..31
      int col = (t & 7) * 4;       // 0..28
#pragma unroll
      for (int p = 0; p < 4; ++p) {
        int r = row + p * 32;
        v4f v = *(const v4f*)(Ag + (size_t)(bm + r) * lda + k0 + col);
        _Float16* dst = sA + r * LDP + col;
        dst[0] = (_Float16)v[0]; dst[1] = (_Float16)v[1];
        dst[2] = (_Float16)v[2]; dst[3] = (_Float16)v[3];
      }
    } else {
      const _Float16* Ag = (const _Float16*)Agv;
      int row = t >> 2;            // 0..63
      int col = (t & 3) * 8;
#pragma unroll
      for (int p = 0; p < 2; ++p) {
        int r = row + p * 64;
        *(v8h*)(sA + r * LDP + col) =
            *(const v8h*)(Ag + (size_t)(bm + r) * lda + k0 + col);
      }
    }
    // ---- stage B tile (BN x BK, N-major) into LDS ----
    if constexpr (!BTRANS) {
      int row = t >> 2;            // 0..63 (N)
      int col = (t & 3) * 8;       // K
      *(v8h*)(sB + row * LDP + col) =
          *(const v8h*)(Bg + (size_t)(bn + row) * ldb + k0 + col);
    } else {
      int kr = t & 31;             // K row
      int n8 = (t >> 5) * 8;       // N group
      v8h v = *(const v8h*)(Bg + (size_t)(k0 + kr) * ldb + bn + n8);
#pragma unroll
      for (int q = 0; q < 8; ++q) sB[(n8 + q) * LDP + kr] = v[q];
    }
    __syncthreads();
    // ---- 2x2 WMMA per wave ----
#pragma unroll
    for (int i = 0; i < 2; ++i) {
      const _Float16* ap = sA + (wm * 32 + i * 16 + m) * LDP;
      // A 16x32 f16 lane layout: elems 0..7 -> K=8*hh.., elems 8..15 -> K=16+8*hh..
      v8h a0 = *(const v8h*)(ap + 8 * hh);
      v8h a1 = *(const v8h*)(ap + 16 + 8 * hh);
      v16h A = __builtin_shufflevector(a0, a1, 0, 1, 2, 3, 4, 5, 6, 7,
                                       8, 9, 10, 11, 12, 13, 14, 15);
#pragma unroll
      for (int j = 0; j < 2; ++j) {
        // B 32x16 f16 lane layout: N=lane&15, elems 0..7 -> K=8*hh.., elems 8..15 -> K=16+8*hh..
        const _Float16* bp = sB + (wn * 32 + j * 16 + m) * LDP;
        v8h b0 = *(const v8h*)(bp + 8 * hh);
        v8h b1 = *(const v8h*)(bp + 16 + 8 * hh);
        v16h B = __builtin_shufflevector(b0, b1, 0, 1, 2, 3, 4, 5, 6, 7,
                                         8, 9, 10, 11, 12, 13, 14, 15);
        c[i][j] = __builtin_amdgcn_wmma_f32_16x16x32_f16(
            false, A, false, B, (short)0, c[i][j], false, false);
      }
    }
  }
}

// epilogue index helper: element vi of c[i][j] lives at
//   row = bm + wm*32 + i*16 + vi + 8*hh ,  col = bn + wn*32 + j*16 + (lane&15)

// ---------------------------------------------------------------------------
// Pack kernels
// ---------------------------------------------------------------------------
__global__ __launch_bounds__(256) void k_pack_im2col(const float* __restrict__ X,
                                                     _Float16* __restrict__ A) {
  int i = blockIdx.x * 256 + threadIdx.x;           // [8192 x 768]
  int r = i / 768, j = i - r * 768;
  int tt = j >> 8, cc = j & 255;
  int s = r >> 3, b = r & 7;
  int sp = s - 2 + tt;
  float v = (sp >= 0) ? X[((size_t)sp * 8 + b) * 256 + cc] : 0.0f;
  A[i] = (_Float16)v;
}

__global__ __launch_bounds__(256) void k_pack_wconv(const float* __restrict__ W,
                                                    _Float16* __restrict__ Wp) {
  int i = blockIdx.x * 256 + threadIdx.x;           // [2048 x 768]
  int o = i / 768, j = i - o * 768;
  int tt = j >> 8, cc = j & 255;
  Wp[i] = (_Float16)W[(size_t)o * 768 + cc * 3 + tt];
}

__global__ __launch_bounds__(256) void k_pack_f16(const float* __restrict__ src,
                                                  _Float16* __restrict__ dst, int n) {
  int i = blockIdx.x * 256 + threadIdx.x;
  if (i < n) dst[i] = (_Float16)src[i];
}

// ---------------------------------------------------------------------------
// Projection GEMM (conv-q/k and v): out packed to [B,H,S,D] f16, +bias
// ---------------------------------------------------------------------------
__global__ __launch_bounds__(256) void k_proj_gemm(const _Float16* __restrict__ A,
                                                   const _Float16* __restrict__ Bw,
                                                   const float* __restrict__ bias,
                                                   _Float16* __restrict__ pack, int K) {
  __shared__ __align__(32) _Float16 sA[BM * LDP];
  __shared__ __align__(32) _Float16 sB[BN * LDP];
  v8f c[2][2] = {{{}, {}}, {{}, {}}};
  int bm = blockIdx.x * BM, bn = blockIdx.y * BN;
  gemm_tile<false, false>(A, K, Bw, K, K, bm, bn, sA, sB, c);
  int lane = threadIdx.x & 31, wave = threadIdx.x >> 5;
  int wm = wave >> 1, wn = wave & 1, m = lane & 15, hh = lane >> 4;
#pragma unroll
  for (int i = 0; i < 2; ++i)
#pragma unroll
    for (int j = 0; j < 2; ++j)
#pragma unroll
      for (int vi = 0; vi < 8; ++vi) {
        int gm = bm + wm * 32 + i * 16 + vi + 8 * hh;   // r = s*8+b
        int gn = bn + wn * 32 + j * 16 + m;             // o = h*256+d
        float val = c[i][j][vi] + bias[gn];
        int s = gm >> 3, b = gm & 7, h = gn >> 8, d = gn & 255;
        pack[(((size_t)(b * 8 + h) * 1024 + s) << 8) + d] = (_Float16)val;
      }
}

// ---------------------------------------------------------------------------
// Scores: attn[z] = q[z] @ k[z]^T * scale + mask   (z = b*8+h)
// ---------------------------------------------------------------------------
__global__ __launch_bounds__(256) void k_scores(const _Float16* __restrict__ qp,
                                                const _Float16* __restrict__ kp,
                                                const float* __restrict__ mask,
                                                float* __restrict__ attn) {
  __shared__ __align__(32) _Float16 sA[BM * LDP];
  __shared__ __align__(32) _Float16 sB[BN * LDP];
  v8f c[2][2] = {{{}, {}}, {{}, {}}};
  int z = blockIdx.z;
  int bm = blockIdx.x * BM, bn = blockIdx.y * BN;
  const _Float16* As = qp + ((size_t)z << 18);
  const _Float16* Bs = kp + ((size_t)z << 18);
  gemm_tile<false, false>(As, 256, Bs, 256, 256, bm, bn, sA, sB, c);
  float* ao = attn + ((size_t)z << 20);
  int lane = threadIdx.x & 31, wave = threadIdx.x >> 5;
  int wm = wave >> 1, wn = wave & 1, m = lane & 15, hh = lane >> 4;
#pragma unroll
  for (int i = 0; i < 2; ++i)
#pragma unroll
    for (int j = 0; j < 2; ++j)
#pragma unroll
      for (int vi = 0; vi < 8; ++vi) {
        int gm = bm + wm * 32 + i * 16 + vi + 8 * hh;   // s
        int gn = bn + wn * 32 + j * 16 + m;             // t
        ao[(size_t)gm * 1024 + gn] = c[i][j][vi] * 0.0625f + mask[(size_t)gm * 1024 + gn];
      }
}

// ---------------------------------------------------------------------------
// Row softmax over 1024 elems, one 256-thread block per row
// ---------------------------------------------------------------------------
__global__ __launch_bounds__(256) void k_softmax(float* __restrict__ attn) {
  __shared__ float red[256];
  float* row = attn + (size_t)blockIdx.x * 1024;
  int t = threadIdx.x;
  v4f v = *(v4f*)(row + t * 4);
  float mx = fmaxf(fmaxf(v[0], v[1]), fmaxf(v[2], v[3]));
  red[t] = mx; __syncthreads();
  for (int s = 128; s > 0; s >>= 1) { if (t < s) red[t] = fmaxf(red[t], red[t + s]); __syncthreads(); }
  mx = red[0]; __syncthreads();
  v[0] = __expf(v[0] - mx); v[1] = __expf(v[1] - mx);
  v[2] = __expf(v[2] - mx); v[3] = __expf(v[3] - mx);
  red[t] = v[0] + v[1] + v[2] + v[3]; __syncthreads();
  for (int s = 128; s > 0; s >>= 1) { if (t < s) red[t] += red[t + s]; __syncthreads(); }
  float inv = 1.0f / red[0];
  v[0] *= inv; v[1] *= inv; v[2] *= inv; v[3] *= inv;
  *(v4f*)(row + t * 4) = v;
}

// ---------------------------------------------------------------------------
// Context: ctx = attn[z] @ v[z]  (A fp32, B transposed-on-load) -> ctx f16 [8192,2048]
// ---------------------------------------------------------------------------
__global__ __launch_bounds__(256) void k_ctx(const float* __restrict__ attn,
                                             const _Float16* __restrict__ vp,
                                             _Float16* __restrict__ ctx) {
  __shared__ __align__(32) _Float16 sA[BM * LDP];
  __shared__ __align__(32) _Float16 sB[BN * LDP];
  v8f c[2][2] = {{{}, {}}, {{}, {}}};
  int z = blockIdx.z;
  int bm = blockIdx.x * BM, bn = blockIdx.y * BN;
  const float* As = attn + ((size_t)z << 20);
  const _Float16* Bs = vp + ((size_t)z << 18);
  gemm_tile<true, true>(As, 1024, Bs, 256, 1024, bm, bn, sA, sB, c);
  int b = z >> 3, h = z & 7;
  int lane = threadIdx.x & 31, wave = threadIdx.x >> 5;
  int wm = wave >> 1, wn = wave & 1, m = lane & 15, hh = lane >> 4;
#pragma unroll
  for (int i = 0; i < 2; ++i)
#pragma unroll
    for (int j = 0; j < 2; ++j)
#pragma unroll
      for (int vi = 0; vi < 8; ++vi) {
        int gm = bm + wm * 32 + i * 16 + vi + 8 * hh;   // s
        int gn = bn + wn * 32 + j * 16 + m;             // d
        ctx[(size_t)(gm * 8 + b) * 2048 + h * 256 + gn] = (_Float16)c[i][j][vi];
      }
}

// ---------------------------------------------------------------------------
// Output projection: out = ctx @ Wo^T + bo   (fp32 out, [S,B,D] == [8192,256])
// ---------------------------------------------------------------------------
__global__ __launch_bounds__(256) void k_out(const _Float16* __restrict__ ctx,
                                             const _Float16* __restrict__ Wop,
                                             const float* __restrict__ bo,
                                             float* __restrict__ out) {
  __shared__ __align__(32) _Float16 sA[BM * LDP];
  __shared__ __align__(32) _Float16 sB[BN * LDP];
  v8f c[2][2] = {{{}, {}}, {{}, {}}};
  int bm = blockIdx.x * BM, bn = blockIdx.y * BN;
  gemm_tile<false, false>(ctx, 2048, Wop, 2048, 2048, bm, bn, sA, sB, c);
  int lane = threadIdx.x & 31, wave = threadIdx.x >> 5;
  int wm = wave >> 1, wn = wave & 1, m = lane & 15, hh = lane >> 4;
#pragma unroll
  for (int i = 0; i < 2; ++i)
#pragma unroll
    for (int j = 0; j < 2; ++j)
#pragma unroll
      for (int vi = 0; vi < 8; ++vi) {
        int gm = bm + wm * 32 + i * 16 + vi + 8 * hh;
        int gn = bn + wn * 32 + j * 16 + m;
        out[(size_t)gm * 256 + gn] = c[i][j][vi] + bo[gn];
      }
}

// ---------------------------------------------------------------------------
extern "C" void kernel_launch(void* const* d_in, const int* in_sizes, int n_in,
                              void* d_out, int out_size, void* d_ws, size_t ws_size,
                              hipStream_t stream) {
  (void)in_sizes; (void)n_in; (void)out_size; (void)ws_size;
  const float* Q    = (const float*)d_in[0];
  const float* Kin  = (const float*)d_in[1];
  const float* V    = (const float*)d_in[2];
  const float* mask = (const float*)d_in[3];
  const float* Wq   = (const float*)d_in[4];
  const float* bq   = (const float*)d_in[5];
  const float* Wk   = (const float*)d_in[6];
  const float* bk   = (const float*)d_in[7];
  const float* Wv   = (const float*)d_in[8];
  const float* bv   = (const float*)d_in[9];
  const float* Wo   = (const float*)d_in[10];
  const float* bo   = (const float*)d_in[11];

  float* out  = (float*)d_out;                 // [8192,256]
  float* attn = out + (size_t)8192 * 256;      // [64,1024,1024] fp32

  // Scratch inside the (not-yet-written) attn region — all consumed before
  // k_scores writes attn; single-stream ordering makes this safe.
  _Float16* Aq  = (_Float16*)attn;                         // [8192,768]
  _Float16* Ak  = Aq  + (size_t)8192 * 768;                // [8192,768]
  _Float16* Wqp = Ak  + (size_t)8192 * 768;                // [2048,768]
  _Float16* Wkp = Wqp + (size_t)2048 * 768;                // [2048,768]
  _Float16* Vh  = Wkp + (size_t)2048 * 768;                // [8192,256]
  _Float16* Wvp = Vh  + (size_t)8192 * 256;                // [2048,256]

  // Persistent scratch in d_ws (~135 MB)
  _Float16* qpack = (_Float16*)d_ws;                       // [64,1024,256]
  _Float16* kpack = qpack + (size_t)64 * 1024 * 256;
  _Float16* vpack = kpack + (size_t)64 * 1024 * 256;
  _Float16* ctx   = vpack + (size_t)64 * 1024 * 256;       // [8192,2048]
  _Float16* Wop   = ctx   + (size_t)8192 * 2048;           // [256,2048]

  // ---- packing ----
  k_pack_im2col<<<24576, 256, 0, stream>>>(Q,   Aq);
  k_pack_im2col<<<24576, 256, 0, stream>>>(Kin, Ak);
  k_pack_wconv <<<6144,  256, 0, stream>>>(Wq, Wqp);
  k_pack_wconv <<<6144,  256, 0, stream>>>(Wk, Wkp);
  k_pack_f16   <<<8192,  256, 0, stream>>>(V,  Vh,  8192 * 256);
  k_pack_f16   <<<2048,  256, 0, stream>>>(Wv, Wvp, 2048 * 256);
  k_pack_f16   <<<2048,  256, 0, stream>>>(Wo, Wop, 256 * 2048);

  // ---- projections (WMMA GEMMs) ----
  k_proj_gemm<<<dim3(64, 32), 256, 0, stream>>>(Aq, Wqp, bq, qpack, 768);
  k_proj_gemm<<<dim3(64, 32), 256, 0, stream>>>(Ak, Wkp, bk, kpack, 768);
  k_proj_gemm<<<dim3(64, 32), 256, 0, stream>>>(Vh, Wvp, bv, vpack, 256);

  // ---- attention ----
  k_scores <<<dim3(8, 16, 64), 256, 0, stream>>>(qpack, kpack, mask, attn);
  k_softmax<<<65536, 256, 0, stream>>>(attn);
  k_ctx    <<<dim3(8, 4, 64), 256, 0, stream>>>(attn, vpack, ctx);

  // ---- output projection ----
  k_out<<<dim3(64, 4), 256, 0, stream>>>(ctx, Wop, bo, out);
}
